// BahdanauAttention_36326833390069
// MI455X (gfx1250) — matrix operand order repaired and checked by reading, not compiled, not run on hardware
//
#include <hip/hip_runtime.h>
#include <hip/hip_bf16.h>

#define HD 1024
#define SD 2048
#define NB 32
#define MT 32          // s-rows per block in k2 (two 16-row WMMA M-tiles)
#define APITCH 1032    // bf16 elems: 2064 B = 516 dwords, 516 % 64 = 4 -> conflict-free

typedef __attribute__((ext_vector_type(16))) __bf16 v16bf;
typedef __attribute__((ext_vector_type(8)))  float  v8f;
typedef __attribute__((ext_vector_type(4)))  float  fx4;
typedef __attribute__((ext_vector_type(4)))  unsigned int uint4v;

union FragA {
    v16bf v;
    struct { uint4v lo, hi; } q;
};

__device__ __forceinline__ unsigned short f2bf(float f) {
    unsigned int u = __float_as_uint(f);
    u += 0x7FFFu + ((u >> 16) & 1u);          // round-to-nearest-even
    return (unsigned short)(u >> 16);
}

__device__ __forceinline__ float fast_tanh(float x) {
#if __has_builtin(__builtin_amdgcn_tanhf)
    return __builtin_amdgcn_tanhf(x);          // v_tanh_f32
#elif __has_builtin(__builtin_amdgcn_tanh_f32)
    return __builtin_amdgcn_tanh_f32(x);       // v_tanh_f32
#else
    return tanhf(x);
#endif
}

// ---------------------------------------------------------------- kernel 0
// Convert Ua_w (1024x1024 f32, row-major [o][h]) to bf16 in workspace.
__global__ __launch_bounds__(256) void k0_convert(const float* __restrict__ Ua,
                                                  unsigned short* __restrict__ Ub) {
    int i = (blockIdx.x * 256 + threadIdx.x) * 4;
    fx4 f = *(const fx4*)(Ua + i);
    Ub[i + 0] = f2bf(f.x);
    Ub[i + 1] = f2bf(f.y);
    Ub[i + 2] = f2bf(f.z);
    Ub[i + 3] = f2bf(f.w);
}

// ---------------------------------------------------------------- kernel 1
// q[b][o] = dot(query[b], Wa_w[o]) + Wa_b[o]   (fp32, tiny)
__global__ __launch_bounds__(256) void k1_qproj(const float* __restrict__ query,
                                                const float* __restrict__ Wa_w,
                                                const float* __restrict__ Wa_b,
                                                float* __restrict__ q) {
    int idx = blockIdx.x * 256 + threadIdx.x;       // 0 .. 32*1024
    int b = idx >> 10, o = idx & 1023;
    const float* qr = query + (size_t)b * HD;
    const float* wr = Wa_w + (size_t)o * HD;
    float acc = 0.f;
    for (int h = 0; h < HD; h += 4) {
        fx4 a = *(const fx4*)(qr + h);
        fx4 c = *(const fx4*)(wr + h);
        acc += a.x * c.x + a.y * c.y + a.z * c.z + a.w * c.w;
    }
    q[idx] = acc + Wa_b[o];
}

// ---------------------------------------------------------------- kernel 2
// Per block: one (batch, 32-row s-tile). Computes
//   scores[b][s] = Va . tanh(q[b] + keys[b][s] @ Ua^T + Ua_b) + Va_b
// via v_wmma_f32_16x16x32_bf16. A (keys, bf16) staged in dynamic LDS,
// each B fragment from L2-resident bf16 Ua is reused by TWO WMMAs (M=32).
__global__ __launch_bounds__(256) void k2_scores(const float* __restrict__ keys,
                                                 const unsigned short* __restrict__ Ub,
                                                 const float* __restrict__ q,
                                                 const float* __restrict__ Ua_b,
                                                 const float* __restrict__ Va_w,
                                                 const float* __restrict__ Va_b,
                                                 float* __restrict__ scores) {
    const int b    = blockIdx.x >> 6;   // / (SD/MT)
    const int st   = blockIdx.x & 63;
    const int s0   = st * MT;
    const int tid  = threadIdx.x;
    const int wave = tid >> 5;
    const int lane = tid & 31;
    const int half = lane >> 4;
    const int ln   = lane & 15;

    extern __shared__ __attribute__((aligned(32))) unsigned short Albs[]; // MT x APITCH bf16
    __shared__ float score_lds[MT];

    if (tid < MT) score_lds[tid] = 0.f;

    // Stage keys tile (32 x 1024 f32) -> bf16 LDS; 256 float4 per row, NT hint.
    const float* krow = keys + ((size_t)b * SD + s0) * HD;
    for (int i = tid; i < MT * 256; i += 256) {
        int r = i >> 8, c4 = i & 255;
        fx4 f = __builtin_nontemporal_load((const fx4*)(krow + (size_t)r * HD + c4 * 4));
        unsigned short* dst = &Albs[r * APITCH + c4 * 4];
        dst[0] = f2bf(f.x); dst[1] = f2bf(f.y); dst[2] = f2bf(f.z); dst[3] = f2bf(f.w);
    }
    __syncthreads();

    // Each wave owns 8 of the 64 N-tiles; two M-tiles (rows 0-15, 16-31).
    float part0[8], part1[8];
#pragma unroll
    for (int j = 0; j < 8; ++j) { part0[j] = 0.f; part1[j] = 0.f; }

    // A fragment bases: row = ln (M-tile 0) and ln+16 (M-tile 1), dword offset 8*half.
    const unsigned short* A0 = Albs + ln * APITCH + half * 16;
    const unsigned short* A1 = Albs + (ln + 16) * APITCH + half * 16;

    for (int t = 0; t < 8; ++t) {
        const int n0 = (wave * 8 + t) * 16;
        v8f acc0 = {};
        v8f acc1 = {};
        // B fragment: lane = column (n0+ln), 16 consecutive K values at k0+16*half
        const unsigned short* Bbase = Ub + (size_t)(n0 + ln) * HD + half * 16;
#pragma unroll 4
        for (int k0 = 0; k0 < HD; k0 += 32) {
            v16bf fb = *(const v16bf*)(Bbase + k0);
            FragA fa0, fa1;
            fa0.q.lo = *(const uint4v*)(A0 + k0);        // dwords 8*half .. +3
            fa0.q.hi = *(const uint4v*)(A0 + k0 + 8);    // dwords 4+8*half .. +3
            fa1.q.lo = *(const uint4v*)(A1 + k0);
            fa1.q.hi = *(const uint4v*)(A1 + k0 + 8);
            acc0 = __builtin_amdgcn_wmma_f32_16x16x32_bf16(
                false, fa0.v, false, fb, (short)0, acc0, false, false);
            acc1 = __builtin_amdgcn_wmma_f32_16x16x32_bf16(
                false, fa1.v, false, fb, (short)0, acc1, false, false);
        }
        // Epilogue: D element (m = j + 8*half, n = ln), o = n0 + ln
        const int o = n0 + ln;
        const float qv = q[b * HD + o] + Ua_b[o];
        const float vv = Va_w[o];
#pragma unroll
        for (int j = 0; j < 8; ++j) {
            part0[j] += fast_tanh(acc0[j] + qv) * vv;
            part1[j] += fast_tanh(acc1[j] + qv) * vv;
        }
    }
#pragma unroll
    for (int j = 0; j < 8; ++j) {
        atomicAdd(&score_lds[j + half * 8], part0[j]);        // ds_add_f32, row m
        atomicAdd(&score_lds[16 + j + half * 8], part1[j]);   // row m+16
    }
    __syncthreads();

    if (tid < MT)
        scores[(size_t)b * SD + s0 + tid] = score_lds[tid] + Va_b[0];
}

// ---------------------------------------------------------------- kernel 3
// Per batch: softmax over 2048 scores -> weights (d_out), then
// context[b][h] = sum_s w[s] * keys[b][s][h]  (coalesced over h).
__global__ __launch_bounds__(1024) void k3_softmax_context(const float* __restrict__ keys,
                                                           const float* __restrict__ scores,
                                                           float* __restrict__ context,
                                                           float* __restrict__ weights) {
    __shared__ float w[SD];
    __shared__ float red[1024];
    const int b = blockIdx.x, t = threadIdx.x;

    float a0 = scores[(size_t)b * SD + t];
    float a1 = scores[(size_t)b * SD + 1024 + t];

    red[t] = fmaxf(a0, a1);
    __syncthreads();
    for (int s = 512; s > 0; s >>= 1) {
        if (t < s) red[t] = fmaxf(red[t], red[t + s]);
        __syncthreads();
    }
    const float mx = red[0];
    __syncthreads();

    float e0 = __expf(a0 - mx), e1 = __expf(a1 - mx);
    red[t] = e0 + e1;
    __syncthreads();
    for (int s = 512; s > 0; s >>= 1) {
        if (t < s) red[t] += red[t + s];
        __syncthreads();
    }
    const float inv = 1.0f / red[0];
    __syncthreads();

    e0 *= inv; e1 *= inv;
    w[t] = e0; w[t + 1024] = e1;
    weights[(size_t)b * SD + t]        = e0;
    weights[(size_t)b * SD + 1024 + t] = e1;
    __syncthreads();

    const float* kb = keys + (size_t)b * SD * HD;
    float acc = 0.f;
#pragma unroll 4
    for (int s = 0; s < SD; ++s)
        acc += w[s] * __builtin_nontemporal_load(kb + (size_t)s * HD + t);
    context[(size_t)b * HD + t] = acc;
}

// ---------------------------------------------------------------- launch
extern "C" void kernel_launch(void* const* d_in, const int* in_sizes, int n_in,
                              void* d_out, int out_size, void* d_ws, size_t ws_size,
                              hipStream_t stream) {
    const float* query = (const float*)d_in[0];   // 32 x 1 x 1024
    const float* keys  = (const float*)d_in[1];   // 32 x 2048 x 1024
    const float* Wa_w  = (const float*)d_in[2];   // 1024 x 1024
    const float* Wa_b  = (const float*)d_in[3];   // 1024
    const float* Ua_w  = (const float*)d_in[4];   // 1024 x 1024
    const float* Ua_b  = (const float*)d_in[5];   // 1024
    const float* Va_w  = (const float*)d_in[6];   // 1 x 1024
    const float* Va_b  = (const float*)d_in[7];   // 1

    float* context = (float*)d_out;                         // 32 x 1024
    float* weights = (float*)d_out + NB * HD;               // 32 x 2048

    char* ws = (char*)d_ws;
    unsigned short* Ub = (unsigned short*)ws;               // 2 MB bf16 Ua
    float* qbuf   = (float*)(ws + (size_t)HD * HD * 2);     // 128 KB
    float* scores = (float*)(ws + (size_t)HD * HD * 2 + (size_t)NB * HD * 4); // 256 KB

    const size_t ldsBytes = (size_t)MT * APITCH * sizeof(unsigned short);  // 66048 B

    k0_convert<<<HD * HD / (256 * 4), 256, 0, stream>>>(Ua_w, Ub);
    k1_qproj<<<NB * HD / 256, 256, 0, stream>>>(query, Wa_w, Wa_b, qbuf);
    k2_scores<<<NB * (SD / MT), 256, ldsBytes, stream>>>(keys, Ub, qbuf, Ua_b, Va_w, Va_b, scores);
    k3_softmax_context<<<NB, 1024, 0, stream>>>(keys, scores, context, weights);
}
